// BaselineMLHAMoELLM_42356967473514
// MI455X (gfx1250) — matrix operand order repaired and checked by reading, not compiled, not run on hardware
//
#include <hip/hip_runtime.h>
#include <math.h>

// ---------------------------------------------------------------------------
// MLA + MoE LLM forward for MI455X (gfx1250, wave32, WMMA bf16).
// - All GEMMs: v_wmma_f32_16x16x32_bf16, f32 accumulate.
// - Weights converted fp32 -> bf16 once per launch (bf16 set ~145MB fits the
//   192MB global L2); MoE weights and V are pre-TRANSPOSED so every GEMM
//   uses the contiguous N x K fragment path (global_load_b128 only).
// - Wave register tile: 32(M) x 64(N) = 8 independent WMMAs per k-step
//   (2 A-frags x 4 B-frags) -> ~4x arithmetic intensity vs 1 tile/wave and
//   enough independent chains to hide WMMA RAW hazards.
// ---------------------------------------------------------------------------

typedef __attribute__((ext_vector_type(16))) __bf16 bf16x16;
typedef __attribute__((ext_vector_type(8)))  float  f32x8;

union FragBF16 {
    uint4 q[2];
    bf16x16 v;
};

#define S_  1024
#define D_  1024
#define H_  16
#define DR_ 64
#define DV_ 64
#define R_  256
#define V_  32000
#define L_  2
#define E_  4
#define F_  4096
#define KVAW_ (R_ + DR_)   // 320
#define SCALE_ 0.125f      // DR^-0.5
#define GEMM_WAVES 8       // waves per block

// ---------------------------------------------------------------------------
// WMMA GEMM:  C[m,n] = act( sum_k A[m,k]*B[n,k] + bias[n] ) * rs[m]
//   A: M x K bf16 row-major (lda);  B: N x K bf16 row-major (ldb).
//   Wave tile: 32x64. Requires M % 32 == 0, N % 64 == 0, K % 32 == 0.
//   Output: Cbf != null -> bf16 store; else f32 store (acc: +=).
// Fragment layouts per CDNA5 ISA (wave32):
//   A 16x32: lane l<16 M=l, K=half*8+{0..7} and 16+half*8+{0..7}
//   B 32x16: lane carries N=l16, K=half*8+{0..7} and 16+half*8+{0..7}
//   C 16x16: vreg i -> M = half*8 + i, N = l16
// ---------------------------------------------------------------------------
__global__ void gemm_bf16_wmma(const __bf16* __restrict__ A, int lda,
                               const __bf16* __restrict__ B, int ldb,
                               float* __restrict__ C, __bf16* __restrict__ Cbf, int ldc,
                               int M, int N, int K,
                               const float* __restrict__ bias,
                               const float* __restrict__ rowscale, int rs_stride,
                               int acc, int gelu)
{
    const int wave = threadIdx.x >> 5;
    const int lane = threadIdx.x & 31;
    const int half = lane >> 4;
    const int l16  = lane & 15;

    const int tiles_n = N >> 6;                       // 64-wide N tiles
    const int tile = blockIdx.x * GEMM_WAVES + wave;  // flattened tile id
    const int tm = tile / tiles_n;
    const int tn = tile - tm * tiles_n;
    if (tm >= (M >> 5)) return;                       // wave-uniform guard
    const int m0 = tm << 5;
    const int n0 = tn << 6;

    f32x8 c[2][4] = {};

    const __bf16* a0 = A + (size_t)(m0 + l16)      * lda + half * 8;
    const __bf16* a1 = A + (size_t)(m0 + 16 + l16) * lda + half * 8;
    const __bf16* b0 = B + (size_t)(n0 + l16)      * ldb + half * 8;

    for (int k = 0; k < K; k += 32) {
        FragBF16 a[2], b[4];
        a[0].q[0] = *reinterpret_cast<const uint4*>(a0 + k);
        a[0].q[1] = *reinterpret_cast<const uint4*>(a0 + k + 16);
        a[1].q[0] = *reinterpret_cast<const uint4*>(a1 + k);
        a[1].q[1] = *reinterpret_cast<const uint4*>(a1 + k + 16);
#pragma unroll
        for (int f = 0; f < 4; ++f) {
            const __bf16* bp = b0 + (size_t)(f * 16) * ldb + k;
            b[f].q[0] = *reinterpret_cast<const uint4*>(bp);
            b[f].q[1] = *reinterpret_cast<const uint4*>(bp + 16);
        }
#pragma unroll
        for (int i = 0; i < 2; ++i)
#pragma unroll
            for (int f = 0; f < 4; ++f)
                c[i][f] = __builtin_amdgcn_wmma_f32_16x16x32_bf16(
                    false, a[i].v, false, b[f].v, (short)0, c[i][f], false, false);
    }

#pragma unroll
    for (int i = 0; i < 2; ++i) {
#pragma unroll
        for (int f = 0; f < 4; ++f) {
            const int col = n0 + f * 16 + l16;
#pragma unroll
            for (int r = 0; r < 8; ++r) {
                const int row = m0 + i * 16 + half * 8 + r;
                float v = c[i][f][r];
                if (bias)     v += bias[col];
                if (gelu)     v = 0.5f * v * (1.0f + erff(v * 0.70710678118f));
                if (rowscale) v *= rowscale[(size_t)row * rs_stride];
                if (Cbf)      Cbf[(size_t)row * ldc + col] = (__bf16)v;
                else if (acc) C[(size_t)row * ldc + col] += v;
                else          C[(size_t)row * ldc + col]  = v;
            }
        }
    }
}

// --------------------------- elementwise helpers ---------------------------
__global__ void cvt_f32_bf16_k(const float* __restrict__ in, __bf16* __restrict__ out,
                               size_t n)
{
    size_t i = (size_t)blockIdx.x * blockDim.x + threadIdx.x;
    const size_t stride = (size_t)gridDim.x * blockDim.x;
    for (; i < n; i += stride) out[i] = (__bf16)in[i];
}

// LDS-tiled transpose + f32->bf16: src rows x cols -> dst cols x rows.
// rows, cols must be multiples of 32. Block 256 = 32x8.
__global__ void transpose_cvt_k(const float* __restrict__ src, __bf16* __restrict__ dst,
                                int rows, int cols)
{
    __shared__ float tile[32][33];
    const int bx = blockIdx.x * 32;          // col base
    const int by = blockIdx.y * 32;          // row base
    const int tx = threadIdx.x & 31;
    const int ty = threadIdx.x >> 5;
#pragma unroll
    for (int i = ty; i < 32; i += 8)
        tile[i][tx] = src[(size_t)(by + i) * cols + bx + tx];
    __syncthreads();
#pragma unroll
    for (int i = ty; i < 32; i += 8)
        dst[(size_t)(bx + i) * rows + by + tx] = (__bf16)tile[tx][i];
}

__global__ void gather_embed_k(const int* __restrict__ tok,
                               const float* __restrict__ embed,
                               float* __restrict__ x)
{
    const int s = blockIdx.x;
    const int r = tok[s];
    for (int i = threadIdx.x; i < D_; i += blockDim.x)
        x[(size_t)s * D_ + i] = embed[(size_t)r * D_ + i];
}

// RMSNorm over W elements of each row (row stride = xstride); writes f32
// and/or bf16 outputs (either may be null).
__global__ void rmsnorm_k(const float* __restrict__ x, int xstride,
                          const float* __restrict__ w,
                          float* __restrict__ of32, __bf16* __restrict__ obf,
                          int ostride, int W)
{
    const int row = blockIdx.x;
    const float* xr = x + (size_t)row * xstride;
    __shared__ float red[8];
    float s = 0.f;
    for (int i = threadIdx.x; i < W; i += blockDim.x) { float t = xr[i]; s += t * t; }
#pragma unroll
    for (int o = 16; o; o >>= 1) s += __shfl_xor(s, o, 32);
    if ((threadIdx.x & 31) == 0) red[threadIdx.x >> 5] = s;
    __syncthreads();
    if (threadIdx.x == 0) {
        float t = 0.f;
        const int nw = blockDim.x >> 5;
        for (int i = 0; i < nw; ++i) t += red[i];
        red[0] = t;
    }
    __syncthreads();
    const float inv = rsqrtf(red[0] / (float)W + 1e-6f);
    for (int i = threadIdx.x; i < W; i += blockDim.x) {
        const float v = xr[i] * inv * w[i];
        if (of32) of32[(size_t)row * ostride + i] = v;
        if (obf)  obf [(size_t)row * ostride + i] = (__bf16)v;
    }
}

__global__ void rope_setup_k(float* __restrict__ cosb, float* __restrict__ sinb)
{
    const int s = blockIdx.x, i = threadIdx.x;  // i in 0..63
    const float inv = __expf(-((float)(i & 31) * 2.0f / (float)DR_) * logf(10000.0f));
    const float f = (float)s * inv;
    cosb[s * DR_ + i] = cosf(f);
    sinb[s * DR_ + i] = sinf(f);
}

// Reference rope: de-interleave pairs then rotate-half. Emits bf16.
__global__ void rope_k(const float* __restrict__ in, int instride, int inoff,
                       __bf16* __restrict__ out, int ostride, int ooff,
                       const float* __restrict__ cosb, const float* __restrict__ sinb)
{
    const int s = blockIdx.x, h = blockIdx.y, j = threadIdx.x; // j < 32
    const float* xr = in + (size_t)s * instride + inoff + h * DR_;
    __bf16* orow = out + (size_t)s * ostride + ooff + h * DR_;
    const float x0 = xr[2 * j], x1 = xr[2 * j + 1];
    const float c0 = cosb[s * DR_ + j],      s0 = sinb[s * DR_ + j];
    const float c1 = cosb[s * DR_ + j + 32], s1 = sinb[s * DR_ + j + 32];
    orow[j]      = (__bf16)(x0 * c0 - x1 * s0);
    orow[j + 32] = (__bf16)(x1 * c1 + x0 * s1);
}

// Fused scale + causal mask + softmax; emits bf16 probabilities for PV GEMM.
__global__ void softmax_causal_k(const float* __restrict__ sc,
                                 __bf16* __restrict__ attn, float scale)
{
    const int row = blockIdx.x;
    const float* r = sc + (size_t)row * S_;
    __shared__ float red[8];
    __shared__ float bc[2];
    const int tid = threadIdx.x;

    float m = -1e30f;
    for (int t = tid; t <= row; t += blockDim.x) m = fmaxf(m, r[t] * scale);
#pragma unroll
    for (int o = 16; o; o >>= 1) m = fmaxf(m, __shfl_xor(m, o, 32));
    if ((tid & 31) == 0) red[tid >> 5] = m;
    __syncthreads();
    if (tid == 0) {
        float mm = red[0];
        for (int i = 1; i < (int)(blockDim.x >> 5); ++i) mm = fmaxf(mm, red[i]);
        bc[0] = mm;
    }
    __syncthreads();
    m = bc[0];

    float s = 0.f;
    for (int t = tid; t <= row; t += blockDim.x) s += __expf(r[t] * scale - m);
#pragma unroll
    for (int o = 16; o; o >>= 1) s += __shfl_xor(s, o, 32);
    if ((tid & 31) == 0) red[tid >> 5] = s;
    __syncthreads();
    if (tid == 0) {
        float ss = 0.f;
        for (int i = 0; i < (int)(blockDim.x >> 5); ++i) ss += red[i];
        bc[1] = 1.0f / ss;
    }
    __syncthreads();
    const float invs = bc[1];

    for (int t = tid; t < S_; t += blockDim.x) {
        const float p = (t <= row) ? __expf(r[t] * scale - m) * invs : 0.0f;
        attn[(size_t)row * S_ + t] = (__bf16)p;
    }
}

// Router: softmax over E=4 logits, top-2 gates, combine weights, aux stats.
__global__ void router_k(const float* __restrict__ h2,
                         const float* __restrict__ rw, const float* __restrict__ rb,
                         float* __restrict__ combine,
                         float* __restrict__ selcnt, float* __restrict__ probsum)
{
    const int t = blockIdx.x;
    const int wave = threadIdx.x >> 5;   // 4 waves = 4 experts
    const int lane = threadIdx.x & 31;
    __shared__ float lg[E_];

    float s = 0.f;
    const float* hr = h2 + (size_t)t * D_;
    const float* wr = rw + (size_t)wave * D_;
    for (int i = lane; i < D_; i += 32) s += hr[i] * wr[i];
#pragma unroll
    for (int o = 16; o; o >>= 1) s += __shfl_xor(s, o, 32);
    if (lane == 0) lg[wave] = s + rb[wave];
    __syncthreads();

    if (threadIdx.x == 0) {
        float mx = lg[0];
        for (int e = 1; e < E_; ++e) mx = fmaxf(mx, lg[e]);
        float p[E_], den = 0.f;
        for (int e = 0; e < E_; ++e) { p[e] = __expf(lg[e] - mx); den += p[e]; }
        for (int e = 0; e < E_; ++e) p[e] /= den;
        int i1 = 0;
        for (int e = 1; e < E_; ++e) if (p[e] > p[i1]) i1 = e;
        int i2 = -1;
        for (int e = 0; e < E_; ++e) if (e != i1 && (i2 < 0 || p[e] > p[i2])) i2 = e;
        const float gs = p[i1] + p[i2];
        for (int e = 0; e < E_; ++e) {
            const float g = (e == i1) ? p[i1] / gs : ((e == i2) ? p[i2] / gs : 0.f);
            combine[(size_t)t * E_ + e] = g;
            atomicAdd(&probsum[e], p[e]);
        }
        atomicAdd(&selcnt[i1], 1.f);
        atomicAdd(&selcnt[i2], 1.f);
    }
}

__global__ void aux_layer_k(const float* __restrict__ selcnt,
                            const float* __restrict__ probsum,
                            float* __restrict__ auxtot)
{
    if (threadIdx.x == 0) {
        float a = 0.f;
        for (int e = 0; e < E_; ++e)
            a += (selcnt[e] / (float)S_) * (probsum[e] / (float)S_);
        auxtot[0] += (float)E_ * a;
    }
}

__global__ void finalize_aux_k(const float* __restrict__ auxtot,
                               float* __restrict__ out)
{
    if (threadIdx.x == 0) out[0] = auxtot[0] * 0.01f;
}

// ---------------------------------------------------------------------------
extern "C" void kernel_launch(void* const* d_in, const int* in_sizes, int n_in,
                              void* d_out, int out_size, void* d_ws, size_t ws_size,
                              hipStream_t stream)
{
    (void)in_sizes; (void)n_in; (void)out_size; (void)ws_size;

    const int*   tokens   = (const int*)  d_in[0];
    const float* embed    = (const float*)d_in[1];
    const float* q_w      = (const float*)d_in[2];
    const float* kva_w    = (const float*)d_in[3];
    const float* kva_ln   = (const float*)d_in[4];
    const float* kvb_w    = (const float*)d_in[5];
    const float* o_w      = (const float*)d_in[6];
    const float* norm1_w  = (const float*)d_in[7];
    const float* norm2_w  = (const float*)d_in[8];
    const float* router_w = (const float*)d_in[9];
    const float* router_b = (const float*)d_in[10];
    const float* w1       = (const float*)d_in[11];
    const float* b1       = (const float*)d_in[12];
    const float* w2       = (const float*)d_in[13];
    const float* b2       = (const float*)d_in[14];
    const float* fnorm_w  = (const float*)d_in[15];
    float* out = (float*)d_out;

    // ---- workspace bump allocator -----------------------------------------
    char* wp = (char*)d_ws;
    auto alloc = [&](size_t bytes) -> void* {
        void* r = (void*)wp;
        wp += (bytes + 255) & ~(size_t)255;
        return r;
    };

    // f32 activation buffers
    float* xbuf    = (float*)alloc((size_t)S_ * D_ * 4);
    float* qbuf    = (float*)alloc((size_t)S_ * H_ * DR_ * 4);
    float* kvabuf  = (float*)alloc((size_t)S_ * KVAW_ * 4);
    float* vbuf    = (float*)alloc((size_t)S_ * H_ * DV_ * 4);
    float* obuf    = (float*)alloc((size_t)S_ * H_ * DV_ * 4);
    float* scores  = (float*)alloc((size_t)S_ * S_ * 4);
    float* h2f     = (float*)alloc((size_t)S_ * D_ * 4);
    float* combine = (float*)alloc((size_t)S_ * E_ * 4);
    float* cosb    = (float*)alloc((size_t)S_ * DR_ * 4);
    float* sinb    = (float*)alloc((size_t)S_ * DR_ * 4);
    float* auxtot  = (float*)alloc(16 * 4);
    float* selcnt  = (float*)alloc(16 * 4);
    float* probsum = (float*)alloc(16 * 4);

    // bf16 activation buffers
    __bf16* hbf    = (__bf16*)alloc((size_t)S_ * D_ * 2);
    __bf16* qbf    = (__bf16*)alloc((size_t)S_ * H_ * DR_ * 2);
    __bf16* kbf    = (__bf16*)alloc((size_t)S_ * DR_ * 2);
    __bf16* ckvbf  = (__bf16*)alloc((size_t)S_ * R_ * 2);
    __bf16* vtbf   = (__bf16*)alloc((size_t)H_ * DV_ * S_ * 2);  // v transposed
    __bf16* obf    = (__bf16*)alloc((size_t)S_ * H_ * DV_ * 2);
    __bf16* attnbf = (__bf16*)alloc((size_t)S_ * S_ * 2);
    __bf16* h2bf   = (__bf16*)alloc((size_t)S_ * D_ * 2);
    __bf16* hmidbf = (__bf16*)alloc((size_t)S_ * F_ * 2);

    // bf16 weight copies (L2-resident working set); w1/w2 transposed to N x K
    __bf16* embedbf = (__bf16*)alloc((size_t)V_ * D_ * 2);
    __bf16* qwbf    = (__bf16*)alloc((size_t)L_ * H_ * DR_ * D_ * 2);
    __bf16* kvawbf  = (__bf16*)alloc((size_t)L_ * KVAW_ * D_ * 2);
    __bf16* kvbwbf  = (__bf16*)alloc((size_t)L_ * H_ * DV_ * R_ * 2);
    __bf16* owbf    = (__bf16*)alloc((size_t)L_ * D_ * H_ * DV_ * 2);
    __bf16* w1tbf   = (__bf16*)alloc((size_t)L_ * E_ * F_ * D_ * 2);  // F x D
    __bf16* w2tbf   = (__bf16*)alloc((size_t)L_ * E_ * D_ * F_ * 2);  // D x F

    auto cvt = [&](const float* src, __bf16* dst, size_t n) {
        unsigned int g = (unsigned int)((n + 255) / 256);
        if (g > 8192u) g = 8192u;
        cvt_f32_bf16_k<<<g, 256, 0, stream>>>(src, dst, n);
    };
    auto tcvt = [&](const float* src, __bf16* dst, int rows, int cols) {
        transpose_cvt_k<<<dim3(cols / 32, rows / 32), 256, 0, stream>>>(src, dst, rows, cols);
    };
    auto gemm = [&](const __bf16* A, int lda, const __bf16* B, int ldb,
                    float* C, __bf16* Cbf, int ldc, int M, int N, int K,
                    const float* bias, const float* rs, int rs_stride,
                    int acc, int gelu) {
        const int tiles = (M >> 5) * (N >> 6);
        gemm_bf16_wmma<<<(tiles + GEMM_WAVES - 1) / GEMM_WAVES, 32 * GEMM_WAVES, 0, stream>>>(
            A, lda, B, ldb, C, Cbf, ldc, M, N, K, bias, rs, rs_stride, acc, gelu);
    };

    // ---- one-time per launch: weights -> bf16 (+transpose), tables --------
    hipMemsetAsync(auxtot, 0, 16 * 4, stream);
    cvt(embed, embedbf, (size_t)V_ * D_);
    cvt(q_w,   qwbf,    (size_t)L_ * H_ * DR_ * D_);
    cvt(kva_w, kvawbf,  (size_t)L_ * KVAW_ * D_);
    cvt(kvb_w, kvbwbf,  (size_t)L_ * H_ * DV_ * R_);
    cvt(o_w,   owbf,    (size_t)L_ * D_ * H_ * DV_);
    for (int le = 0; le < L_ * E_; ++le) {
        tcvt(w1 + (size_t)le * D_ * F_, w1tbf + (size_t)le * F_ * D_, D_, F_); // -> F x D
        tcvt(w2 + (size_t)le * F_ * D_, w2tbf + (size_t)le * D_ * F_, F_, D_); // -> D x F
    }
    rope_setup_k<<<S_, DR_, 0, stream>>>(cosb, sinb);

    // ---- embedding --------------------------------------------------------
    gather_embed_k<<<S_, 256, 0, stream>>>(tokens, embed, xbuf);

    // ---- transformer layers ----------------------------------------------
    for (int l = 0; l < L_; ++l) {
        rmsnorm_k<<<S_, 256, 0, stream>>>(xbuf, D_, norm1_w + (size_t)l * D_,
                                          nullptr, hbf, D_, D_);
        // q = h @ q_w^T ; kva = h @ kva_w^T
        gemm(hbf, D_, qwbf + (size_t)l * H_ * DR_ * D_, D_,
             qbuf, nullptr, H_ * DR_, S_, H_ * DR_, D_, nullptr, nullptr, 0, 0, 0);
        gemm(hbf, D_, kvawbf + (size_t)l * KVAW_ * D_, D_,
             kvabuf, nullptr, KVAW_, S_, KVAW_, D_, nullptr, nullptr, 0, 0, 0);
        // rope
        rope_k<<<dim3(S_, H_), 32, 0, stream>>>(qbuf, H_ * DR_, 0, qbf, H_ * DR_, 0, cosb, sinb);
        rope_k<<<dim3(S_, 1), 32, 0, stream>>>(kvabuf, KVAW_, R_, kbf, DR_, 0, cosb, sinb);
        // ckv = rms_norm(kva[:, :R]) ; v = ckv @ kvb_w^T ; vT for PV GEMM
        rmsnorm_k<<<S_, 256, 0, stream>>>(kvabuf, KVAW_, kva_ln + (size_t)l * R_,
                                          nullptr, ckvbf, R_, R_);
        gemm(ckvbf, R_, kvbwbf + (size_t)l * H_ * DV_ * R_, R_,
             vbuf, nullptr, H_ * DV_, S_, H_ * DV_, R_, nullptr, nullptr, 0, 0, 0);
        tcvt(vbuf, vtbf, S_, H_ * DV_);   // (H*DV) x S, bf16

        // attention per head (scores: 4MB f32, L2-resident)
        for (int h = 0; h < H_; ++h) {
            gemm(qbf + h * DR_, H_ * DR_, kbf, DR_,
                 scores, nullptr, S_, S_, S_, DR_, nullptr, nullptr, 0, 0, 0);
            softmax_causal_k<<<S_, 256, 0, stream>>>(scores, attnbf, SCALE_);
            gemm(attnbf, S_, vtbf + (size_t)h * DV_ * S_, S_,
                 obuf + h * DV_, nullptr, H_ * DV_, S_, DV_, S_,
                 nullptr, nullptr, 0, 0, 0);
        }
        cvt(obuf, obf, (size_t)S_ * H_ * DV_);
        // x += o @ o_w^T
        gemm(obf, H_ * DV_, owbf + (size_t)l * D_ * H_ * DV_, H_ * DV_,
             xbuf, nullptr, D_, S_, D_, H_ * DV_, nullptr, nullptr, 0, 1, 0);

        // h2 = rms_norm(x, norm2)
        rmsnorm_k<<<S_, 256, 0, stream>>>(xbuf, D_, norm2_w + (size_t)l * D_,
                                          h2f, h2bf, D_, D_);

        // router + aux stats
        hipMemsetAsync(selcnt, 0, 32 * 4, stream);
        router_k<<<S_, 128, 0, stream>>>(h2f, router_w + (size_t)l * E_ * D_,
                                         router_b + (size_t)l * E_,
                                         combine, selcnt, probsum);
        aux_layer_k<<<1, 32, 0, stream>>>(selcnt, probsum, auxtot);

        // MoE (dense, matching reference): fused gate + bias + residual
        for (int e = 0; e < E_; ++e) {
            const size_t we = ((size_t)l * E_ + e);
            gemm(h2bf, D_, w1tbf + we * F_ * D_, D_,
                 nullptr, hmidbf, F_, S_, F_, D_,
                 b1 + we * F_, nullptr, 0, 0, 1);
            gemm(hmidbf, F_, w2tbf + we * D_ * F_, F_,
                 xbuf, nullptr, D_, S_, D_, F_,
                 b2 + we * D_, combine + e, E_, 1, 0);
        }
    }

    // ---- final norm + logits ---------------------------------------------
    rmsnorm_k<<<S_, 256, 0, stream>>>(xbuf, D_, fnorm_w, nullptr, hbf, D_, D_);
    gemm(hbf, D_, embedbf, D_,
         out, nullptr, V_, S_, V_, D_, nullptr, nullptr, 0, 0, 0);
    finalize_aux_k<<<1, 32, 0, stream>>>(auxtot, out + (size_t)S_ * V_);
}
